// EncoderSelfAttention_89180700934477
// MI455X (gfx1250) — matrix operand order repaired and checked by reading, not compiled, not run on hardware
//
#include <hip/hip_runtime.h>
#include <hip/hip_bf16.h>

// ---------------------------------------------------------------------------
// Types for CDNA5 WMMA (wave32): v_wmma_f32_16x16x32_bf16
// ---------------------------------------------------------------------------
typedef __attribute__((ext_vector_type(16))) __bf16 v16bf;
typedef __attribute__((ext_vector_type(8)))  __bf16 v8bf;
typedef __attribute__((ext_vector_type(8)))  float  v8f;

#define BATCH 2
#define SEQ   2048
#define DMODEL 1024
#define NHEAD 16
#define DHEAD 64
#define FMIN  (-3.402823466e38f)
#define SCALE (0.03125f)   /* 1/sqrt(1024) */

static __device__ __forceinline__ unsigned short f2bf(float f) {
    unsigned int u = __float_as_uint(f);
    unsigned int r = u + 0x7FFFu + ((u >> 16) & 1u);   // round-to-nearest-even
    return (unsigned short)(r >> 16);
}

static __device__ __forceinline__ v8f wmma_bf16(v16bf a, v16bf b, v8f c) {
    return __builtin_amdgcn_wmma_f32_16x16x32_bf16(false, a, false, b,
                                                   (short)0, c, false, false);
}

// A fragment (16x32 bf16, row m = lane%16): per lane 16 contiguous K values at
// K = 16*half  -> two contiguous 16B loads.
static __device__ __forceinline__ v16bf load_a_frag(const unsigned short* rowp,
                                                    int koff, int half) {
    v8bf lo = *(const v8bf*)(rowp + koff + 16 * half);
    v8bf hi = *(const v8bf*)(rowp + koff + 16 * half + 8);
    return __builtin_shufflevector(lo, hi, 0,1,2,3,4,5,6,7,8,9,10,11,12,13,14,15);
}

// B fragment (32x16 bf16, col n = lane%16): per lane 16 contiguous K values at
// K = 16*half, loaded from B^T (N x K row-major) -> one contiguous 32B load.
static __device__ __forceinline__ v16bf load_b_frag(const unsigned short* btRow,
                                                    int half) {
    return *(const v16bf*)(btRow + 16 * half);
}

// ---------------------------------------------------------------------------
// fp32 -> bf16 elementwise
// ---------------------------------------------------------------------------
__global__ __launch_bounds__(256) void k_cvt_bf16(const float* __restrict__ src,
                                                  unsigned short* __restrict__ dst,
                                                  int n) {
    int i = blockIdx.x * blockDim.x + threadIdx.x;
    int stride = gridDim.x * blockDim.x;
    for (; i < n; i += stride) dst[i] = f2bf(src[i]);
}

// fp32 (K x N) -> bf16 transposed (N x K)
__global__ __launch_bounds__(256) void k_cvt_T(const float* __restrict__ src,
                                               unsigned short* __restrict__ dst,
                                               int K, int N) {
    int idx = blockIdx.x * blockDim.x + threadIdx.x;
    if (idx >= K * N) return;
    int k = idx / N, n = idx % N;
    dst[(size_t)n * K + k] = f2bf(src[(size_t)k * N + n]);
}

// V flat (B*L x 1024 bf16, [pos][h*64+d]) -> VT (B,H,64,L bf16)
__global__ __launch_bounds__(256) void k_vtrans(const unsigned short* __restrict__ v,
                                                unsigned short* __restrict__ vt) {
    int idx = blockIdx.x * blockDim.x + threadIdx.x;   // dst-linear
    int l  = idx & (SEQ - 1);
    int t  = idx >> 11;
    int d  = t & (DHEAD - 1);
    int t2 = t >> 6;
    int h  = t2 & (NHEAD - 1);
    int b  = t2 >> 4;
    vt[idx] = v[((size_t)(b * SEQ + l)) * DMODEL + h * DHEAD + d];
}

// ---------------------------------------------------------------------------
// bf16 WMMA GEMM: C(MxN) = A(MxK) @ B(KxN),  A row-major bf16,
// BT = B^T (NxK) row-major bf16.
// One wave -> 32x64 output tile: 8 accumulators, two A row-groups share the
// same four B fragments each k-step (8 wmma per 2+4 b128-load-pairs).
// ---------------------------------------------------------------------------
__global__ __launch_bounds__(256) void k_gemm_wmma(const unsigned short* __restrict__ A,
                                                   const unsigned short* __restrict__ BT,
                                                   void* __restrict__ C,
                                                   int M, int N, int K, int f32out) {
    int lane = threadIdx.x & 31;
    int wid  = threadIdx.x >> 5;
    int w    = blockIdx.x * 8 + wid;
    int tilesN = N >> 6;
    int tm = w / tilesN;
    int tn = w - tm * tilesN;
    if (tm * 32 >= M) return;
    int ln = lane & 15, half = lane >> 4;

    const unsigned short* arow0 = A + (size_t)(tm * 32 + ln) * K;
    const unsigned short* arow1 = A + (size_t)(tm * 32 + 16 + ln) * K;
    v8f acc[8];
#pragma unroll
    for (int j = 0; j < 8; ++j)
#pragma unroll
        for (int r = 0; r < 8; ++r) acc[j][r] = 0.0f;

    for (int kk = 0; kk < K; kk += 32) {
        // cover latency one k-tile ahead (global_prefetch_b8)
        if (kk + 32 < K) {
            __builtin_prefetch(arow0 + kk + 32, 0, 3);
            __builtin_prefetch(arow1 + kk + 32, 0, 3);
        }
        v16bf a0 = load_a_frag(arow0, kk, half);
        v16bf a1 = load_a_frag(arow1, kk, half);
#pragma unroll
        for (int j = 0; j < 4; ++j) {
            const unsigned short* btRow =
                BT + (size_t)(tn * 64 + 16 * j + ln) * K + kk;
            v16bf b = load_b_frag(btRow, half);
            acc[j]     = wmma_bf16(a0, b, acc[j]);
            acc[4 + j] = wmma_bf16(a1, b, acc[4 + j]);
        }
    }

#pragma unroll
    for (int g = 0; g < 2; ++g)
#pragma unroll
        for (int j = 0; j < 4; ++j)
#pragma unroll
            for (int r = 0; r < 8; ++r) {
                int m = tm * 32 + g * 16 + r + 8 * half;  // C layout: m=r+8*half
                size_t o = (size_t)m * N + tn * 64 + 16 * j + ln;
                float vv = acc[g * 4 + j][r];
                if (f32out) ((float*)C)[o] = vv;
                else        ((unsigned short*)C)[o] = f2bf(vv);
            }
}

// ---------------------------------------------------------------------------
// Flash-style attention. One wave = one (b, h, 16-row q tile).
// Qf/Kf: (B*L x 1024) bf16 flat; VT: (B,H,64,L) bf16; CTX out: (B*L x 1024) bf16.
// Faithful to reference: logit += FLT_MIN * pmask, THEN * (1/32), softmax.
// ---------------------------------------------------------------------------
__global__ __launch_bounds__(128) void k_attn(const unsigned short* __restrict__ Qf,
                                              const unsigned short* __restrict__ Kf,
                                              const unsigned short* __restrict__ VT,
                                              const int* __restrict__ amask,
                                              unsigned short* __restrict__ CTX) {
    __shared__ __align__(64) unsigned short lds[4][16 * 32];   // P staging / wave
    int lane = threadIdx.x & 31;
    int wid  = threadIdx.x >> 5;
    int w    = blockIdx.x * 4 + wid;
    int b    = w >> 11;                 // 16 heads * 128 q-tiles
    int rem  = w & 2047;
    int h    = rem >> 7;
    int q0   = (rem & 127) << 4;
    int ln = lane & 15, half = lane >> 4;

    // Q fragments for this 16-row tile (dh = 64 -> two K=32 fragments), hoisted.
    const unsigned short* qrow =
        Qf + (size_t)(b * SEQ + q0 + ln) * DMODEL + h * DHEAD;
    v16bf qa0 = load_a_frag(qrow, 0, half);
    v16bf qa1 = load_a_frag(qrow, 32, half);

    float qm[8], rmax[8], rsum[8];
#pragma unroll
    for (int r = 0; r < 8; ++r) {
        qm[r]   = (float)amask[b * SEQ + q0 + r + 8 * half];
        rmax[r] = -__builtin_inff();
        rsum[r] = 0.0f;
    }
    v8f acc[4];
#pragma unroll
    for (int j = 0; j < 4; ++j)
#pragma unroll
        for (int r = 0; r < 8; ++r) acc[j][r] = 0.0f;

    unsigned short* Pl = lds[wid];
    const unsigned short* vbase0 = VT + (size_t)((b * NHEAD + h) * DHEAD) * SEQ;

    for (int kc = 0; kc < SEQ; kc += 32) {
        // ---- S = Q Kt for a 16x32 chunk of logits (4 wmma) ----
        const unsigned short* kbase =
            Kf + (size_t)(b * SEQ + kc) * DMODEL + h * DHEAD + 16 * half;
        if (kc + 32 < SEQ) {
            __builtin_prefetch(kbase + (size_t)32 * DMODEL, 0, 3);
            __builtin_prefetch(kbase + (size_t)48 * DMODEL, 0, 3);
        }
        v8f s0, s1;
#pragma unroll
        for (int r = 0; r < 8; ++r) { s0[r] = 0.0f; s1[r] = 0.0f; }
        v16bf kb;
        kb = *(const v16bf*)(kbase + (size_t)(0  + ln) * DMODEL + 0);
        s0 = wmma_bf16(qa0, kb, s0);
        kb = *(const v16bf*)(kbase + (size_t)(0  + ln) * DMODEL + 32);
        s0 = wmma_bf16(qa1, kb, s0);
        kb = *(const v16bf*)(kbase + (size_t)(16 + ln) * DMODEL + 0);
        s1 = wmma_bf16(qa0, kb, s1);
        kb = *(const v16bf*)(kbase + (size_t)(16 + ln) * DMODEL + 32);
        s1 = wmma_bf16(qa1, kb, s1);

        // ---- mask + scale (faithful order), chunk row-max ----
        float km0 = (float)amask[b * SEQ + kc + ln];
        float km1 = (float)amask[b * SEQ + kc + 16 + ln];
        float cmax[8];
#pragma unroll
        for (int r = 0; r < 8; ++r) {
            s0[r] = (s0[r] + FMIN * (1.0f - qm[r] * km0)) * SCALE;
            s1[r] = (s1[r] + FMIN * (1.0f - qm[r] * km1)) * SCALE;
            cmax[r] = fmaxf(s0[r], s1[r]);
        }
#pragma unroll
        for (int off = 1; off < 16; off <<= 1)
#pragma unroll
            for (int r = 0; r < 8; ++r)
                cmax[r] = fmaxf(cmax[r], __shfl_xor(cmax[r], off, 32));

        // ---- online softmax update ----
        float fs[8], psum[8];
#pragma unroll
        for (int r = 0; r < 8; ++r) {
            float nm = fmaxf(rmax[r], cmax[r]);
            fs[r] = __expf(rmax[r] - nm);
            rmax[r] = nm;
            float p0 = __expf(s0[r] - nm);
            float p1 = __expf(s1[r] - nm);
            s0[r] = p0; s1[r] = p1;
            psum[r] = p0 + p1;
        }
#pragma unroll
        for (int off = 1; off < 16; off <<= 1)
#pragma unroll
            for (int r = 0; r < 8; ++r)
                psum[r] += __shfl_xor(psum[r], off, 32);
#pragma unroll
        for (int r = 0; r < 8; ++r) rsum[r] = rsum[r] * fs[r] + psum[r];
#pragma unroll
        for (int j = 0; j < 4; ++j)
#pragma unroll
            for (int r = 0; r < 8; ++r) acc[j][r] *= fs[r];

        // ---- stage P (C layout) -> LDS -> reload as A fragment ----
#pragma unroll
        for (int r = 0; r < 8; ++r) {
            Pl[(r + 8 * half) * 32 + ln]      = f2bf(s0[r]);
            Pl[(r + 8 * half) * 32 + 16 + ln] = f2bf(s1[r]);
        }
        asm volatile("s_wait_dscnt 0" ::: "memory");   // intra-wave LDS RAW
        __builtin_amdgcn_wave_barrier();
        v16bf pa = load_a_frag(Pl + ln * 32, 0, half);
        __builtin_amdgcn_wave_barrier();

        // ---- ctx += P @ V_chunk (4 wmma over dh=64) ----
#pragma unroll
        for (int j = 0; j < 4; ++j) {
            v16bf vb = *(const v16bf*)(vbase0 + (size_t)(16 * j + ln) * SEQ +
                                       kc + 16 * half);
            acc[j] = wmma_bf16(pa, vb, acc[j]);
        }
    }

    // ---- normalize + store ctx in concat layout [pos][h*64+d] ----
#pragma unroll
    for (int j = 0; j < 4; ++j)
#pragma unroll
        for (int r = 0; r < 8; ++r) {
            size_t o = (size_t)(b * SEQ + q0 + r + 8 * half) * DMODEL +
                       h * DHEAD + 16 * j + ln;
            CTX[o] = f2bf(acc[j][r] / rsum[r]);
        }
}

// ---------------------------------------------------------------------------
extern "C" void kernel_launch(void* const* d_in, const int* in_sizes, int n_in,
                              void* d_out, int out_size, void* d_ws, size_t ws_size,
                              hipStream_t stream) {
    const float* x  = (const float*)d_in[0];
    const int*   am = (const int*)d_in[1];
    const float* wq = (const float*)d_in[2];
    const float* wk = (const float*)d_in[3];
    const float* wv = (const float*)d_in[4];
    const float* wo = (const float*)d_in[5];

    char* ws = (char*)d_ws;
    const size_t SZ_X  = (size_t)BATCH * SEQ * DMODEL * 2;   // bf16 bytes
    const size_t SZ_W  = (size_t)DMODEL * DMODEL * 2;
    unsigned short* xb  = (unsigned short*)(ws);
    unsigned short* wqT = (unsigned short*)(ws + SZ_X);
    unsigned short* wkT = (unsigned short*)(ws + SZ_X + SZ_W);
    unsigned short* wvT = (unsigned short*)(ws + SZ_X + 2 * SZ_W);
    unsigned short* woT = (unsigned short*)(ws + SZ_X + 3 * SZ_W);
    unsigned short* Qf  = (unsigned short*)(ws + SZ_X + 4 * SZ_W);
    unsigned short* Kf  = (unsigned short*)(ws + 2 * SZ_X + 4 * SZ_W);
    unsigned short* Vf  = (unsigned short*)(ws + 3 * SZ_X + 4 * SZ_W);
    unsigned short* VT  = (unsigned short*)(ws + 4 * SZ_X + 4 * SZ_W);
    unsigned short* CTX = (unsigned short*)(ws + 5 * SZ_X + 4 * SZ_W);

    const int NX = BATCH * SEQ * DMODEL;           // 4,194,304
    k_cvt_bf16<<<4096, 256, 0, stream>>>(x, xb, NX);
    k_cvt_T<<<(DMODEL * DMODEL) / 256, 256, 0, stream>>>(wq, wqT, DMODEL, DMODEL);
    k_cvt_T<<<(DMODEL * DMODEL) / 256, 256, 0, stream>>>(wk, wkT, DMODEL, DMODEL);
    k_cvt_T<<<(DMODEL * DMODEL) / 256, 256, 0, stream>>>(wv, wvT, DMODEL, DMODEL);
    k_cvt_T<<<(DMODEL * DMODEL) / 256, 256, 0, stream>>>(wo, woT, DMODEL, DMODEL);

    const int M = BATCH * SEQ;                      // 4096
    const int gemmBlocks = (M / 32) * (DMODEL / 64) / 8;   // 256
    k_gemm_wmma<<<gemmBlocks, 256, 0, stream>>>(xb, wqT, Qf, M, DMODEL, DMODEL, 0);
    k_gemm_wmma<<<gemmBlocks, 256, 0, stream>>>(xb, wkT, Kf, M, DMODEL, DMODEL, 0);
    k_gemm_wmma<<<gemmBlocks, 256, 0, stream>>>(xb, wvT, Vf, M, DMODEL, DMODEL, 0);

    k_vtrans<<<NX / 256, 256, 0, stream>>>(Vf, VT);

    const int attnBlocks = BATCH * NHEAD * (SEQ / 16) / 4;  // 1024
    k_attn<<<attnBlocks, 128, 0, stream>>>(Qf, Kf, VT, am, CTX);

    k_gemm_wmma<<<gemmBlocks, 256, 0, stream>>>(CTX, woT, d_out, M, DMODEL, DMODEL, 1);
}
